// Transformer_66288525246591
// MI455X (gfx1250) — matrix operand order repaired and checked by reading, not compiled, not run on hardware
//
#include <hip/hip_runtime.h>
#include <hip/hip_bf16.h>
#include <math.h>

// Model constants (match reference)
#define Vv  1024
#define Dd  128
#define Hh  8
#define Ll  2
#define Ss  1024
#define Bb  8
#define HDd 16
#define Mrows (Bb * Ss)   // 8192
#define NC   8            // scan chunks per (b,h)
#define CL   (Ss / NC)    // 128 steps per chunk

typedef __attribute__((ext_vector_type(16))) _Float16 v16h;
typedef __attribute__((ext_vector_type(8)))  _Float16 h8;
typedef __attribute__((ext_vector_type(8)))  float    v8f;
typedef __attribute__((ext_vector_type(4)))  unsigned u32x4;
typedef __attribute__((ext_vector_type(4)))  int      i32x4;
typedef __attribute__((ext_vector_type(8)))  int      i32x8;

#if __has_builtin(__builtin_amdgcn_tensor_load_to_lds)
#define HAVE_TDM 1
#endif

// Two contiguous 16B loads -> one 16-half WMMA fragment
__device__ __forceinline__ v16h load_frag16(const _Float16* p0, const _Float16* p1)
{
    union { v16h v; h8 p[2]; } u;
    u.p[0] = *reinterpret_cast<const h8*>(p0);
    u.p[1] = *reinterpret_cast<const h8*>(p1);
    return u.v;
}

// ---------------------------------------------------------------------------
// TDM: load one f32 column (stride H) of the [S,H] filter into LDS.
// D# per ISA §8.3-8.5: group0 = {count/lds_addr/global_addr/type=2},
// group1 = {data_size=4B, tensor_dim0=H, tensor_dim1=S, tile=1xS, stride0=H}.
// ---------------------------------------------------------------------------
__device__ __forceinline__ void tdm_load_filter_col(const float* gcol, unsigned lds_byte_off)
{
#ifdef HAVE_TDM
    unsigned long long ga = (unsigned long long)(size_t)gcol;
    u32x4 g0 = {};
    g0[0] = 1u;                                    // count=1, user descriptor
    g0[1] = lds_byte_off;                          // lds_addr
    g0[2] = (unsigned)ga;                          // global_addr[31:0]
    g0[3] = (unsigned)((ga >> 32) & 0x01ffffffu) | (2u << 30);  // addr[56:32] | type=2
    i32x8 g1 = {};
    g1[0] = (2 << 16);                             // data_size code 2 = 4 bytes
    g1[1] = (Hh & 0xffff) << 16;                   // tensor_dim0 = H
    g1[2] = (int)(((unsigned)Ss & 0xffffu) << 16); // tensor_dim1 = S (low16)
    g1[3] = (1 << 16);                             // tile_dim0 = 1
    g1[4] = Ss;                                    // tile_dim1 = S
    g1[5] = Hh;                                    // tensor_dim0_stride = H
    i32x4 gz = {};
#if __clang_major__ >= 23
    i32x8 gz8 = {};
    __builtin_amdgcn_tensor_load_to_lds(g0, g1, gz, gz, gz8, 0);
#else
    __builtin_amdgcn_tensor_load_to_lds(g0, g1, gz, gz, 0);
#endif
#endif
}

// ---------------------------------------------------------------------------
// Embedding + sinusoidal positional encoding (computed on the fly)
// ---------------------------------------------------------------------------
__global__ void embed_kernel(const int* __restrict__ tokens,
                             const float* __restrict__ emb,
                             float* __restrict__ X)
{
    int idx = blockIdx.x * blockDim.x + threadIdx.x;
    if (idx >= Mrows * Dd) return;
    int dcol = idx & (Dd - 1);
    int bs   = idx >> 7;
    int s    = bs & (Ss - 1);
    int tok  = tokens[bs];
    float ang = -(float)(dcol & ~1) * (9.210340371976184f / (float)Dd);
    float dv  = __expf(ang);
    float arg = (float)s * dv;
    float pe  = (dcol & 1) ? __cosf(arg) : __sinf(arg);
    X[idx] = emb[(size_t)tok * Dd + dcol] + pe;
}

// ---------------------------------------------------------------------------
// Weight prep: W [K,N] f32 row-major  ->  Wt [N,K] f16 (transposed)
// ---------------------------------------------------------------------------
__global__ void wt_prep(const float* __restrict__ W, _Float16* __restrict__ Wt,
                        int K, int N)
{
    int idx = blockIdx.x * blockDim.x + threadIdx.x;
    if (idx >= K * N) return;
    int k = idx / N, n = idx - k * N;
    Wt[(size_t)n * K + k] = (_Float16)W[idx];
}

// ---------------------------------------------------------------------------
// LayerNorm over D=128: one wave per row; f32 in, f16 out (feeds GEMMs only)
// ---------------------------------------------------------------------------
__global__ __launch_bounds__(256) void ln_kernel(const float* __restrict__ X,
                                                 const float* __restrict__ gam,
                                                 const float* __restrict__ bet,
                                                 _Float16* __restrict__ Y)
{
    int row  = blockIdx.x * 8 + (threadIdx.x >> 5);
    int lane = threadIdx.x & 31;
    const float* xp = X + (size_t)row * Dd;
    float v[4];
    float m = 0.f;
#pragma unroll
    for (int j = 0; j < 4; ++j) { v[j] = xp[lane + 32 * j]; m += v[j]; }
#pragma unroll
    for (int off = 16; off > 0; off >>= 1) m += __shfl_xor(m, off, 32);
    m *= (1.f / Dd);
    float var = 0.f;
#pragma unroll
    for (int j = 0; j < 4; ++j) { float d = v[j] - m; var += d * d; }
#pragma unroll
    for (int off = 16; off > 0; off >>= 1) var += __shfl_xor(var, off, 32);
    var *= (1.f / Dd);
    float inv = rsqrtf(var + 1e-6f);
    _Float16* yp = Y + (size_t)row * Dd;
#pragma unroll
    for (int j = 0; j < 4; ++j) {
        int c = lane + 32 * j;
        yp[c] = (_Float16)((v[j] - m) * inv * gam[c] + bet[c]);
    }
}

// ---------------------------------------------------------------------------
// WMMA GEMM: out = act((A @ Wt^T + bias) * scale) [+ resid]
// A: [M,K] f16 row-major; Wt: [N,K] f16 (pre-transposed weights).
// outmode: 0 f32 [M,N] (+resid), 1 f32 [B,H,S,HD], 2 f16 [M,N], 3 f16 [B,H,HD,S]
// ---------------------------------------------------------------------------
__global__ __launch_bounds__(256) void gemm16(const _Float16* __restrict__ A,
                                              const _Float16* __restrict__ Wt,
                                              const float* __restrict__ bias,
                                              const float* __restrict__ resid,
                                              void* __restrict__ Cout,
                                              int M, int N, int K,
                                              float scale, int act, int outmode)
{
    const int wave = threadIdx.x >> 5;
    const int lane = threadIdx.x & 31;
    const int half = lane >> 4;
    const int r    = lane & 15;
    const int ntn  = N >> 4;
    int tile = blockIdx.x * 8 + wave;
    if (tile >= (M >> 4) * ntn) return;
    int tm = tile / ntn;
    int tn = tile - tm * ntn;
    const int m0 = tm << 4, n0 = tn << 4;

    const _Float16* aRow = A  + (size_t)(m0 + r) * K;
    const _Float16* bRow = Wt + (size_t)(n0 + r) * K;

    v8f acc = {};
#pragma unroll 4
    for (int k0 = 0; k0 < K; k0 += 32) {
        v16h a = load_frag16(aRow + k0 + half * 8,  aRow + k0 + half * 8 + 16);
        v16h b = load_frag16(bRow + k0 + half * 8,  bRow + k0 + half * 8 + 16);
        acc = __builtin_amdgcn_wmma_f32_16x16x32_f16(false, a, false, b,
                                                     (short)0, acc, false, false);
    }
    const float bval = bias ? bias[n0 + r] : 0.f;
#pragma unroll
    for (int j = 0; j < 8; ++j) {
        int m = m0 + half * 8 + j;
        int n = n0 + r;
        float vv = (acc[j] + bval) * scale;
        if (act == 1) {  // tanh-approx GELU
            float x3 = vv * vv * vv;
            vv = 0.5f * vv * (1.f + tanhf(0.7978845608028654f * (vv + 0.044715f * x3)));
        }
        if (outmode == 0) {
            if (resid) vv += resid[(size_t)m * N + n];
            ((float*)Cout)[(size_t)m * N + n] = vv;
        } else if (outmode == 1) {           // f32 [b,h,s,hd]
            int bb = m >> 10, s = m & (Ss - 1);
            int hh = n >> 4,  hd = n & 15;
            ((float*)Cout)[(((size_t)(bb * Hh + hh) * Ss + s) * HDd) + hd] = vv;
        } else if (outmode == 2) {           // f16 [M,N]
            ((_Float16*)Cout)[(size_t)m * N + n] = (_Float16)vv;
        } else {                             // f16 [b,h,hd,s] (time-major for conv)
            int bb = m >> 10, s = m & (Ss - 1);
            int hh = n >> 4,  hd = n & 15;
            ((_Float16*)Cout)[(((size_t)(bb * Hh + hh) * HDd + hd) * Ss) + s] = (_Float16)vv;
        }
    }
}

// ---------------------------------------------------------------------------
// Causal conv as lower-triangular Toeplitz matmul on WMMA.
// y[s,d] = sum_{t<=s} f[h,s-t] x[t,d].  X f16 [B,H,HD,S]; Y f32 [B,H,S,HD].
// Filter column fetched by the Tensor Data Mover into LDS, converted to f16.
// ---------------------------------------------------------------------------
__global__ __launch_bounds__(256) void conv_wmma(const _Float16* __restrict__ X,
                                                 const float* __restrict__ filt, // [S,H]
                                                 float* __restrict__ Y)
{
    __shared__ _Float16 fsh[Ss];
    const int bh = blockIdx.x;
    const int h  = bh & (Hh - 1);
#ifdef HAVE_TDM
    __shared__ float fstage[Ss];
    if (threadIdx.x < 32) {
        tdm_load_filter_col(filt + h, (unsigned)(size_t)fstage);
#if __has_builtin(__builtin_amdgcn_s_wait_tensorcnt)
        __builtin_amdgcn_s_wait_tensorcnt(0);
#endif
    }
    __syncthreads();
    for (int t = threadIdx.x; t < Ss; t += 256) fsh[t] = (_Float16)fstage[t];
#else
    for (int t = threadIdx.x; t < Ss; t += 256) fsh[t] = (_Float16)filt[(size_t)t * Hh + h];
#endif
    __syncthreads();

    const int wave = threadIdx.x >> 5, lane = threadIdx.x & 31;
    const int half = lane >> 4, r = lane & 15;
    const int s0 = blockIdx.y * 128 + wave * 16;
    const int s  = s0 + r;
    const _Float16* xp = X + ((size_t)bh * HDd + r) * Ss;  // row = channel d=r

    v8f acc = {};
#pragma unroll 2
    for (int t0 = 0; t0 < s0 + 16; t0 += 32) {
        v16h a, b;
#pragma unroll
        for (int i = 0; i < 16; ++i) {
            int t = t0 + half * 8 + i + ((i & 8) ? 8 : 0);
            a[i] = (t <= s) ? fsh[s - t] : (_Float16)0.0f;   // causal mask
        }
        b = load_frag16(xp + t0 + half * 8, xp + t0 + half * 8 + 16);
        acc = __builtin_amdgcn_wmma_f32_16x16x32_f16(false, a, false, b,
                                                     (short)0, acc, false, false);
    }
    float* yp = Y + (size_t)bh * Ss * HDd;
#pragma unroll
    for (int j = 0; j < 8; ++j)
        yp[(size_t)(s0 + half * 8 + j) * HDd + r] = acc[j];
}

// ---------------------------------------------------------------------------
// Gate: g[b,h,s] = relu(sum_{d,e} v_d k_e gk[d*16+e] + gb)^2 + EPS
// ---------------------------------------------------------------------------
__global__ __launch_bounds__(256) void gate_kernel(const float* __restrict__ kc,
                                                   const float* __restrict__ vc,
                                                   const float* __restrict__ gk,
                                                   const float* __restrict__ gb,
                                                   float* __restrict__ gates)
{
    int wid  = blockIdx.x * 8 + (threadIdx.x >> 5);
    int lane = threadIdx.x & 31;
    int s  = wid & (Ss - 1);
    int bh = wid >> 10;
    const float* kp = kc + ((size_t)bh * Ss + s) * HDd;
    const float* vp = vc + ((size_t)bh * Ss + s) * HDd;
    float sum = 0.f;
#pragma unroll
    for (int j = 0; j < 8; ++j) {
        int idx = lane + 32 * j;                  // idx = d*16 + e
        sum += vp[idx >> 4] * kp[idx & 15] * gk[idx];
    }
#pragma unroll
    for (int off = 16; off > 0; off >>= 1) sum += __shfl_xor(sum, off, 32);
    if (lane == 0) {
        float gl = sum + gb[0];
        float rl = gl > 0.f ? gl : 0.f;
        gates[(size_t)bh * Ss + s] = rl * rl + 1e-5f;
    }
}

// ---------------------------------------------------------------------------
// Chunked scan, phase 1: per-chunk totals of g*Z (256 channels) and g.
// Grid (B*H, NC); thread owns channel (d = t&15, e = t>>4).
// ---------------------------------------------------------------------------
__global__ __launch_bounds__(256) void scan_partial(const float* __restrict__ kc,
                                                    const float* __restrict__ vc,
                                                    const float* __restrict__ gates,
                                                    float* __restrict__ Zc,  // [BH,NC,256]
                                                    float* __restrict__ Gc)  // [BH,NC]
{
    const int bh = blockIdx.x, c = blockIdx.y;
    const int t  = threadIdx.x;
    const int d  = t & 15, e = t >> 4;
    const float* kp = kc + ((size_t)bh * Ss + c * CL) * HDd;
    const float* vp = vc + ((size_t)bh * Ss + c * CL) * HDd;
    const float* gp = gates + (size_t)bh * Ss + c * CL;
    float accZ = 0.f, accG = 0.f;
    for (int s = 0; s < CL; ++s) {
        float g = gp[s];
        accZ += g * vp[(size_t)s * HDd + d] * kp[(size_t)s * HDd + e];
        accG += g;
    }
    Zc[((size_t)bh * NC + c) * 256 + t] = accZ;
    if (t == 0) Gc[(size_t)bh * NC + c] = accG;
}

// ---------------------------------------------------------------------------
// Chunked scan, phase 2: fold exclusive chunk prefix, run local scan,
// contract with q, reduce over d (16-lane-group xor shuffle), write o.
// ---------------------------------------------------------------------------
__global__ __launch_bounds__(256) void scan_apply(const float* __restrict__ q,
                                                  const float* __restrict__ kc,
                                                  const float* __restrict__ vc,
                                                  const float* __restrict__ gates,
                                                  const float* __restrict__ Zc,
                                                  const float* __restrict__ Gc,
                                                  float* __restrict__ o)
{
    const int bh   = blockIdx.x, c = blockIdx.y;
    const int t    = threadIdx.x;
    const int lane = t & 31;
    const int d    = t & 15;
    const int e    = t >> 4;

    float accZ = 0.f, accG = 0.f;                  // exclusive prefix over chunks
    for (int cc = 0; cc < c; ++cc) {
        accZ += Zc[((size_t)bh * NC + cc) * 256 + t];
        accG += Gc[(size_t)bh * NC + cc];
    }

    const size_t base = ((size_t)bh * Ss + c * CL) * HDd;
    const float* kp = kc + base;
    const float* vp = vc + base;
    const float* qp = q  + base;
    const float* gp = gates + (size_t)bh * Ss + c * CL;
    float* op = o + base;

    for (int s = 0; s < CL; ++s) {
        if (s + 8 < CL) {                          // global_prefetch_b8
            __builtin_prefetch(kp + (size_t)(s + 8) * HDd, 0, 0);
            __builtin_prefetch(vp + (size_t)(s + 8) * HDd, 0, 0);
        }
        float g = gp[s];
        accZ += g * vp[(size_t)s * HDd + d] * kp[(size_t)s * HDd + e];
        accG += g;
        float attn = accZ / (accG + 1e-5f);
        float val  = qp[(size_t)s * HDd + d] * attn;
#pragma unroll
        for (int off = 1; off < 16; off <<= 1)
            val += __shfl_xor(val, off, 32);
        if ((lane & 15) == 0)
            op[(size_t)s * HDd + e] = val;
    }
}

// ---------------------------------------------------------------------------
// Normalize o over HD, transpose [B,H,S,HD] -> [B,S,D], emit f16 for wo GEMM
// ---------------------------------------------------------------------------
__global__ void norm_transpose(const float* __restrict__ o, _Float16* __restrict__ ho)
{
    int idx = blockIdx.x * blockDim.x + threadIdx.x;  // over B*H*S
    if (idx >= Bb * Hh * Ss) return;
    int s  = idx & (Ss - 1);
    int bh = idx >> 10;
    int b  = bh >> 3, h = bh & 7;
    const float* op = o + (size_t)idx * HDd;
    float ssum = 0.f;
#pragma unroll
    for (int j = 0; j < HDd; ++j) ssum += op[j] * op[j];
    float inv = 1.f / fmaxf(sqrtf(ssum), 1e-5f);
    _Float16* hp = ho + ((size_t)b * Ss + s) * Dd + h * HDd;
#pragma unroll
    for (int j = 0; j < HDd; ++j) hp[j] = (_Float16)(op[j] * inv);
}

// ---------------------------------------------------------------------------
// Launch
// ---------------------------------------------------------------------------
extern "C" void kernel_launch(void* const* d_in, const int* in_sizes, int n_in,
                              void* d_out, int out_size, void* d_ws, size_t ws_size,
                              hipStream_t stream)
{
    const int*   tokens = (const int*)  d_in[0];
    const float* sb     = (const float*)d_in[1];
    const float* emb    = (const float*)d_in[2];
    const float* ln1_s  = (const float*)d_in[3];
    const float* ln1_b  = (const float*)d_in[4];
    const float* wq_k   = (const float*)d_in[5];
    const float* wq_b   = (const float*)d_in[6];
    const float* wk_k   = (const float*)d_in[7];
    const float* wk_b   = (const float*)d_in[8];
    const float* wv_k   = (const float*)d_in[9];
    const float* wv_b   = (const float*)d_in[10];
    const float* wo_k   = (const float*)d_in[11];
    const float* wo_b   = (const float*)d_in[12];
    const float* gate_k = (const float*)d_in[13];
    const float* gate_b = (const float*)d_in[14];
    const float* ln2_s  = (const float*)d_in[15];
    const float* ln2_b  = (const float*)d_in[16];
    const float* ff1_k  = (const float*)d_in[17];
    const float* ff1_b  = (const float*)d_in[18];
    const float* ff2_k  = (const float*)d_in[19];
    const float* ff2_b  = (const float*)d_in[20];
    const float* lnf_s  = (const float*)d_in[21];
    const float* lnf_b  = (const float*)d_in[22];
    const float* out_k  = (const float*)d_in[23];
    const float* out_b  = (const float*)d_in[24];

    const size_t MD = (size_t)Mrows * Dd;       // 1M elements
    const size_t BH = (size_t)Bb * Hh;

    // f32 scratch
    float* f32p  = (float*)d_ws;
    float* x     = f32p;  f32p += MD;           // [M,128]
    float* q     = f32p;  f32p += MD;           // [B,H,S,HD]
    float* kc    = f32p;  f32p += MD;           // [B,H,S,HD]
    float* vc    = f32p;  f32p += MD;           // [B,H,S,HD]
    float* obuf  = f32p;  f32p += MD;           // [B,H,S,HD]
    float* gts   = f32p;  f32p += BH * Ss;      // [BH,S]
    float* Zc    = f32p;  f32p += BH * NC * 256;// chunk Z totals
    float* Gc    = f32p;  f32p += BH * NC;      // chunk gate totals
    f32p += 256 - ((BH * NC) & 255);            // realign to 1KB-ish boundary

    // f16 scratch
    _Float16* f16p = (_Float16*)f32p;
    _Float16* h16  = f16p;  f16p += MD;              // LN outputs [M,128]
    _Float16* ho16 = f16p;  f16p += MD;              // attn out   [M,128]
    _Float16* ff16 = f16p;  f16p += (size_t)Mrows * 4 * Dd;  // [M,512]
    _Float16* k16  = f16p;  f16p += MD;              // [B,H,HD,S]
    _Float16* v16  = f16p;  f16p += MD;              // [B,H,HD,S]
    _Float16* wtb  = f16p;                           // transposed f16 weights
    const size_t WQ = (size_t)Dd * Dd;               // 16384
    const size_t WF = (size_t)Dd * 4 * Dd;           // 65536
    const size_t WLAYER = 4 * WQ + 2 * WF;           // per-layer halfs
    _Float16* outk16 = wtb + Ll * WLAYER;            // [V, D] transposed

    dim3 b256(256);
    const int M = Mrows;
    auto gemmGrid = [](int M_, int N_) { return ((M_ >> 4) * (N_ >> 4) + 7) / 8; };
    auto prepGrid = [](size_t n) { return (unsigned)((n + 255) / 256); };

    // --- weight prep (f32 [K,N] -> f16 [N,K]) ---
    for (int l = 0; l < Ll; ++l) {
        _Float16* wl = wtb + l * WLAYER;
        wt_prep<<<prepGrid(WQ), b256, 0, stream>>>(wq_k + l * WQ, wl + 0 * WQ, Dd, Dd);
        wt_prep<<<prepGrid(WQ), b256, 0, stream>>>(wk_k + l * WQ, wl + 1 * WQ, Dd, Dd);
        wt_prep<<<prepGrid(WQ), b256, 0, stream>>>(wv_k + l * WQ, wl + 2 * WQ, Dd, Dd);
        wt_prep<<<prepGrid(WQ), b256, 0, stream>>>(wo_k + l * WQ, wl + 3 * WQ, Dd, Dd);
        wt_prep<<<prepGrid(WF), b256, 0, stream>>>(ff1_k + l * WF, wl + 4 * WQ, Dd, 4 * Dd);
        wt_prep<<<prepGrid(WF), b256, 0, stream>>>(ff2_k + l * WF, wl + 4 * WQ + WF, 4 * Dd, Dd);
    }
    wt_prep<<<prepGrid((size_t)Dd * Vv), b256, 0, stream>>>(out_k, outk16, Dd, Vv);

    embed_kernel<<<(M * Dd + 255) / 256, b256, 0, stream>>>(tokens, emb, x);

    for (int l = 0; l < Ll; ++l) {
        _Float16* wl = wtb + l * WLAYER;
        ln_kernel<<<M / 8, b256, 0, stream>>>(x, ln1_s + l * Dd, ln1_b + l * Dd, h16);
        gemm16<<<gemmGrid(M, Dd), b256, 0, stream>>>(h16, wl + 0 * WQ, wq_b + l * Dd,
            nullptr, q,   M, Dd, Dd, 1.f,   0, 1);   // f32 head layout
        gemm16<<<gemmGrid(M, Dd), b256, 0, stream>>>(h16, wl + 1 * WQ, wk_b + l * Dd,
            nullptr, k16, M, Dd, Dd, 0.25f, 0, 3);   // f16 time-major (1/sqrt(HD))
        gemm16<<<gemmGrid(M, Dd), b256, 0, stream>>>(h16, wl + 2 * WQ, wv_b + l * Dd,
            nullptr, v16, M, Dd, Dd, 1.f,   0, 3);

        conv_wmma<<<dim3(Bb * Hh, Ss / 128), b256, 0, stream>>>(k16, sb, kc);
        conv_wmma<<<dim3(Bb * Hh, Ss / 128), b256, 0, stream>>>(v16, sb, vc);

        gate_kernel<<<Bb * Hh * Ss / 8, b256, 0, stream>>>(kc, vc,
            gate_k + l * HDd * HDd, gate_b + l, gts);
        scan_partial<<<dim3(Bb * Hh, NC), b256, 0, stream>>>(kc, vc, gts, Zc, Gc);
        scan_apply  <<<dim3(Bb * Hh, NC), b256, 0, stream>>>(q, kc, vc, gts, Zc, Gc, obuf);
        norm_transpose<<<(Bb * Hh * Ss + 255) / 256, b256, 0, stream>>>(obuf, ho16);

        gemm16<<<gemmGrid(M, Dd), b256, 0, stream>>>(ho16, wl + 3 * WQ, wo_b + l * Dd,
            x, x, M, Dd, Dd, 1.f, 0, 0);             // + residual

        ln_kernel<<<M / 8, b256, 0, stream>>>(x, ln2_s + l * Dd, ln2_b + l * Dd, h16);
        gemm16<<<gemmGrid(M, 4 * Dd), b256, 0, stream>>>(h16, wl + 4 * WQ,
            ff1_b + l * 4 * Dd, nullptr, ff16, M, 4 * Dd, Dd, 1.f, 1 /*gelu*/, 2);
        gemm16<<<gemmGrid(M, Dd), b256, 0, stream>>>(ff16, wl + 4 * WQ + WF,
            ff2_b + l * Dd, x, x, M, Dd, 4 * Dd, 1.f, 0, 0);     // + residual
    }

    ln_kernel<<<M / 8, b256, 0, stream>>>(x, lnf_s, lnf_b, h16);
    gemm16<<<gemmGrid(M, Vv), b256, 0, stream>>>(h16, outk16, out_b, nullptr,
        (float*)d_out, M, Vv, Dd, 1.f, 0, 0);
}
